// MoleculeGNSWrapper_20048907338194
// MI455X (gfx1250) — compile-verified, matrix operand order checked by reading, not run on hardware
//
#include <hip/hip_runtime.h>
#include <hip/hip_bf16.h>
#include <math.h>

typedef __bf16 bf16_t;
typedef __attribute__((ext_vector_type(16))) __bf16 v16bf;
typedef __attribute__((ext_vector_type(8)))  __bf16 v8bf;
typedef __attribute__((ext_vector_type(8)))  float  v8f;
typedef __attribute__((ext_vector_type(4)))  int    v4i;

#define LDS_LD 40   // padded LDS row stride (bf16): 80B, 16B aligned, bank-skewed

// ---- gfx1250 async global->LDS path (guarded; falls back to VGPR staging) ----
#if defined(__gfx1250__) && __has_builtin(__builtin_amdgcn_global_load_async_to_lds_b128)
#define USE_ASYNC_LDS 1
#else
#define USE_ASYNC_LDS 0
#endif

#if USE_ASYNC_LDS
typedef __attribute__((address_space(1))) v4i* g_v4i_p;   // global ptr to int4
typedef __attribute__((address_space(3))) v4i* l_v4i_p;   // LDS ptr to int4

__device__ __forceinline__ void async_cp16(const bf16_t* g, bf16_t* l) {
  __builtin_amdgcn_global_load_async_to_lds_b128((g_v4i_p)g, (l_v4i_p)l, 0, 0);
}
__device__ __forceinline__ void async_wait4() {
#if __has_builtin(__builtin_amdgcn_s_wait_asynccnt)
  __builtin_amdgcn_s_wait_asynccnt(4);
#else
  asm volatile("s_wait_asynccnt 4" ::: "memory");
#endif
}
__device__ __forceinline__ void async_wait0() {
#if __has_builtin(__builtin_amdgcn_s_wait_asynccnt)
  __builtin_amdgcn_s_wait_asynccnt(0);
#else
  asm volatile("s_wait_asynccnt 0" ::: "memory");
#endif
}
#endif

// ---------------------------------------------------------------------------
// Generic gather-GEMM:  C[M x Ntot] = concat_K(A0|A1|A2) @ Wt^T + bias
//   - A segments bf16 row-major; optional per-row int gather index
//   - Wt pre-transposed [Ntot][Ktot] bf16
//   - epilogue: bias, optional silu, optional RMS (Ntot==128 only),
//     optional outA = val and/or outB = resB + val
// ---------------------------------------------------------------------------
struct GemmP {
  const bf16_t* a0; const int* i0; int ld0; int k0;
  const bf16_t* a1; const int* i1; int ld1; int k1;
  const bf16_t* a2; const int* i2; int ld2; int k2;
  const bf16_t* wt;            // [Ntot][Ktot]
  const float*  bias;          // [Ntot]
  bf16_t* outA; int ldoA;      // optional
  bf16_t* outB; const bf16_t* resB; int ldoB;  // optional: outB = resB + val
  int M; int Ntot; int flags;  // flags: 1 = silu, 2 = rms
};

__global__ __launch_bounds__(256)
void gemm_kernel(GemmP p) {
#if USE_ASYNC_LDS
  __shared__ bf16_t sA[2][128 * LDS_LD];
  __shared__ bf16_t sB[2][128 * LDS_LD];
#else
  __shared__ bf16_t sA[1][128 * LDS_LD];
  __shared__ bf16_t sB[1][128 * LDS_LD];
#endif

  const int tid  = threadIdx.x;
  const int wave = tid >> 5;
  const int lane = tid & 31;
  const int half = lane >> 4;
  const int l15  = lane & 15;
  const int tileM = blockIdx.x * 128;
  const int cb    = blockIdx.y;
  const int Ktot  = p.k0 + p.k1 + p.k2;

  // per-thread load slots: two rows (rl0, rl1), one 8-elem chunk each
  const int rl0  = tid >> 2;
  const int rl1  = 64 + (tid >> 2);
  const int colq = (tid & 3) * 8;
  const int lofs0 = rl0 * LDS_LD + colq;
  const int lofs1 = rl1 * LDS_LD + colq;
  const bf16_t* wb0 = p.wt + (long)(cb * 128 + rl0) * Ktot + colq;
  const bf16_t* wb1 = p.wt + (long)(cb * 128 + rl1) * Ktot + colq;

  v8f acc[8];
  #pragma unroll
  for (int t = 0; t < 8; ++t)
    #pragma unroll
    for (int i = 0; i < 8; ++i) acc[t][i] = 0.0f;

  const int kc0 = half ? 8 : 0;
  const int kc1 = half ? 24 : 16;
  const int am  = wave * 16 + l15;

  auto compute = [&](const bf16_t* bA, const bf16_t* bB) {
    v8bf alo = *(const v8bf*)(bA + am * LDS_LD + kc0);
    v8bf ahi = *(const v8bf*)(bA + am * LDS_LD + kc1);
    v16bf af;
    #pragma unroll
    for (int i = 0; i < 8; ++i) { af[i] = alo[i]; af[i + 8] = ahi[i]; }
    #pragma unroll
    for (int t = 0; t < 8; ++t) {
      const bf16_t* bp = bB + (t * 16 + l15) * LDS_LD;
      v8bf blo = *(const v8bf*)(bp + kc0);
      v8bf bhi = *(const v8bf*)(bp + kc1);
      v16bf bfv;
      #pragma unroll
      for (int i = 0; i < 8; ++i) { bfv[i] = blo[i]; bfv[i + 8] = bhi[i]; }
      acc[t] = __builtin_amdgcn_wmma_f32_16x16x32_bf16(
          false, af, false, bfv, (short)0, acc[t], false, false);
    }
  };

#if USE_ASYNC_LDS
  int par = 0;
#endif

  // One loop per concat segment: gather index + base pointers hoisted out of
  // the K loop (no per-tile segment branching).  Row index clamped to M-1:
  // garbage rows only affect unstored output rows.
  auto run_seg = [&](const bf16_t* seg, const int* idx, int ld, int kbeg, int kseg) {
    if (kseg <= 0) return;
    int g0 = tileM + rl0; if (g0 >= p.M) g0 = p.M - 1;
    int g1 = tileM + rl1; if (g1 >= p.M) g1 = p.M - 1;
    const long r0 = idx ? (long)idx[g0] : (long)g0;
    const long r1 = idx ? (long)idx[g1] : (long)g1;
    const bf16_t* s0 = seg + r0 * (long)ld + colq;
    const bf16_t* s1 = seg + r1 * (long)ld + colq;

#if USE_ASYNC_LDS
    // double-buffered async copy: issue tile t+1, wait asynccnt<=4 (in-order
    // completion => tile t batch landed), barrier, WMMA on tile t.
    async_cp16(s0,         sA[par] + lofs0);
    async_cp16(s1,         sA[par] + lofs1);
    async_cp16(wb0 + kbeg, sB[par] + lofs0);
    async_cp16(wb1 + kbeg, sB[par] + lofs1);
    for (int kk = 0; kk < kseg; kk += 32) {
      const bool more = (kk + 32) < kseg;
      if (more) {
        const int kt = kbeg + kk + 32;
        const int nb = par ^ 1;
        async_cp16(s0 + kk + 32, sA[nb] + lofs0);
        async_cp16(s1 + kk + 32, sA[nb] + lofs1);
        async_cp16(wb0 + kt,     sB[nb] + lofs0);
        async_cp16(wb1 + kt,     sB[nb] + lofs1);
        async_wait4();
      } else {
        async_wait0();
      }
      __syncthreads();
      compute(sA[par], sB[par]);
      __syncthreads();
      par ^= 1;
    }
#else
    for (int kk = 0; kk < kseg; kk += 32) {
      const int kt = kbeg + kk;
      *(v8bf*)(sA[0] + lofs0) = *(const v8bf*)(s0 + kk);
      *(v8bf*)(sA[0] + lofs1) = *(const v8bf*)(s1 + kk);
      *(v8bf*)(sB[0] + lofs0) = *(const v8bf*)(wb0 + kt);
      *(v8bf*)(sB[0] + lofs1) = *(const v8bf*)(wb1 + kt);
      __builtin_prefetch((const void*)(s0 + kk + 32), 0, 0);
      __builtin_prefetch((const void*)(s1 + kk + 32), 0, 0);
      __syncthreads();
      compute(sA[0], sB[0]);
      __syncthreads();
    }
#endif
  };

  run_seg(p.a0, p.i0, p.ld0, 0, p.k0);
  run_seg(p.a1, p.i1, p.ld1, p.k0, p.k1);
  run_seg(p.a2, p.i2, p.ld2, p.k0 + p.k1, p.k2);

  // ---- epilogue: bias (+silu) ----
  #pragma unroll
  for (int t = 0; t < 8; ++t) {
    const float bv = p.bias ? p.bias[cb * 128 + t * 16 + l15] : 0.0f;
    #pragma unroll
    for (int r = 0; r < 8; ++r) {
      float x = acc[t][r] + bv;
      if (p.flags & 1) x = x / (1.0f + __expf(-x));
      acc[t][r] = x;
    }
  }

  // ---- RMS over the 128-col row (Ntot==128 launches only) ----
  if (p.flags & 2) {
    #pragma unroll
    for (int r = 0; r < 8; ++r) {
      float s = 0.0f;
      #pragma unroll
      for (int t = 0; t < 8; ++t) s += acc[t][r] * acc[t][r];
      s += __shfl_xor(s, 1); s += __shfl_xor(s, 2);
      s += __shfl_xor(s, 4); s += __shfl_xor(s, 8);
      const float sc = rsqrtf(s * (1.0f / 128.0f) + 1e-6f);
      #pragma unroll
      for (int t = 0; t < 8; ++t) acc[t][r] *= sc;
    }
  }

  // ---- stores (row M = r + 8*half within the wave's 16-row stripe) ----
  if (p.outA) {
    #pragma unroll
    for (int r = 0; r < 8; ++r) {
      const int row = tileM + wave * 16 + half * 8 + r;
      if (row >= p.M) continue;
      bf16_t* op = p.outA + (long)row * p.ldoA + cb * 128 + l15;
      #pragma unroll
      for (int t = 0; t < 8; ++t) op[t * 16] = (bf16_t)acc[t][r];
    }
  }
  if (p.outB) {
    #pragma unroll
    for (int r = 0; r < 8; ++r) {
      const int row = tileM + wave * 16 + half * 8 + r;
      if (row >= p.M) continue;
      const long base = (long)row * p.ldoB + cb * 128 + l15;
      #pragma unroll
      for (int t = 0; t < 8; ++t) {
        const float rb = p.resB ? (float)p.resB[base + t * 16] : 0.0f;
        p.outB[base + t * 16] = (bf16_t)(rb + acc[t][r]);
      }
    }
  }
}

// ---------------------------------------------------------------------------
// Weight convert+transpose: W[K x N] f32 -> Wt[N x Kpad] bf16 (zero padded)
// ---------------------------------------------------------------------------
__global__ void wt_conv_kernel(const float* w, bf16_t* wt, int K, int Ncols, int Kpad) {
  const long i = (long)blockIdx.x * blockDim.x + threadIdx.x;
  if (i >= (long)Ncols * Kpad) return;
  const int n = (int)(i / Kpad), k = (int)(i % Kpad);
  wt[i] = (bf16_t)((k < K) ? w[(long)k * Ncols + n] : 0.0f);
}

__global__ void f2bf_kernel(const float* src, bf16_t* dst, long n) {
  const long i = (long)blockIdx.x * blockDim.x + threadIdx.x;
  if (i < n) dst[i] = (bf16_t)src[i];
}

// ---------------------------------------------------------------------------
// Node embedding concat -> bf16 [N x 128]
// ---------------------------------------------------------------------------
__global__ void node_embed_kernel(const int* at, const int* ac, const int* rc, const int* ri,
                                  const float* eat, const float* eac, const float* erc,
                                  const float* eri, bf16_t* out, int N) {
  const long i = (long)blockIdx.x * blockDim.x + threadIdx.x;
  if (i >= (long)N * 128) return;
  const int node = (int)(i >> 7), c = (int)(i & 127);
  float v;
  if (c < 32)       v = eat[at[node] * 32 + c];
  else if (c < 64)  v = eac[ac[node] * 32 + (c - 32)];
  else if (c < 96)  v = erc[rc[node] * 32 + (c - 64)];
  else              v = eri[ri[node] * 32 + (c - 96)];
  out[i] = (bf16_t)v;
}

// ---------------------------------------------------------------------------
// Edge geometry: rbf x ylm x cutoff (72) ++ bond_emb (16), pad to 96; + cut[E]
// ---------------------------------------------------------------------------
__global__ void edge_geom_kernel(const float* pos, const float* c_in,
                                 const int* snd, const int* rcv, const int* bond,
                                 const float* emb_bond, bf16_t* feat, float* cut, int E) {
  const int e = blockIdx.x * blockDim.x + threadIdx.x;
  if (e >= E) return;
  const float ci = c_in[0];
  const int si = snd[e], ri = rcv[e];
  const float vx = (pos[si * 3 + 0] - pos[ri * 3 + 0]) / ci;
  const float vy = (pos[si * 3 + 1] - pos[ri * 3 + 1]) / ci;
  const float vz = (pos[si * 3 + 2] - pos[ri * 3 + 2]) / ci;
  const float rr  = sqrtf(vx * vx + vy * vy + vz * vz + 1e-12f);
  const float inv = 1.0f / rr;
  const float x = vx * inv, y = vy * inv, z = vz * inv;
  const float PI = 3.14159265358979f;
  const float ct = (rr < 5.0f) ? 0.5f * (cosf(PI * rr * 0.2f) + 1.0f) : 0.0f;
  cut[e] = ct;
  const float s3 = sqrtf(3.0f), s5 = sqrtf(5.0f), s15 = sqrtf(15.0f);
  float sh[9];
  sh[0] = 1.0f;        sh[1] = s3 * x;      sh[2] = s3 * y;  sh[3] = s3 * z;
  sh[4] = s15 * x * y; sh[5] = s15 * y * z; sh[6] = 0.5f * s5 * (3.0f * z * z - 1.0f);
  sh[7] = s15 * x * z; sh[8] = 0.5f * s15 * (x * x - y * y);
  const float amp = sqrtf(2.0f / 5.0f);
  bf16_t* fp = feat + (long)e * 96;
  #pragma unroll
  for (int b = 0; b < 8; ++b) {
    const float rbf = amp * __sinf((float)(b + 1) * PI * rr * 0.2f) * inv;
    const float rc2 = rbf * ct;
    #pragma unroll
    for (int l = 0; l < 9; ++l) fp[b * 9 + l] = (bf16_t)(rc2 * sh[l]);
  }
  const float* be = emb_bond + bond[e] * 16;
  #pragma unroll
  for (int j = 0; j < 16; ++j) fp[72 + j] = (bf16_t)be[j];
  #pragma unroll
  for (int j = 88; j < 96; ++j) fp[j] = (bf16_t)0.0f;
}

// ---------------------------------------------------------------------------
// Gate + scatter-add: one wave per edge, fp32 atomics into agg[N x 128]
// ---------------------------------------------------------------------------
__global__ __launch_bounds__(256)
void gate_scatter_kernel(const bf16_t* m, const int* rcv, const float* cut,
                         const float* gw, const float* gb, float* agg, int E) {
  const int wave = threadIdx.x >> 5, lane = threadIdx.x & 31;
  const int e = blockIdx.x * 8 + wave;
  if (e >= E) return;
  const bf16_t* mp = m + (long)e * 128;
  float mv[4]; float dot = 0.0f;
  #pragma unroll
  for (int j = 0; j < 4; ++j) {
    const int c = lane + j * 32;
    mv[j] = (float)mp[c];
    dot += mv[j] * gw[c];
  }
  dot += __shfl_xor(dot, 16); dot += __shfl_xor(dot, 8);
  dot += __shfl_xor(dot, 4);  dot += __shfl_xor(dot, 2); dot += __shfl_xor(dot, 1);
  const float g = (1.0f / (1.0f + __expf(-(dot + gb[0])))) * cut[e];
  float* ap = agg + (long)rcv[e] * 128;
  #pragma unroll
  for (int j = 0; j < 4; ++j) atomicAdd(ap + lane + j * 32, mv[j] * g);
}

// ---------------------------------------------------------------------------
// Head layer 2: [N x 256] bf16 @ [256 x 3] f32 + b -> f32 out
// ---------------------------------------------------------------------------
__global__ void head2_kernel(const bf16_t* hid, const float* w2, const float* b2,
                             float* out, int N) {
  const int node = blockIdx.x * blockDim.x + threadIdx.x;
  if (node >= N) return;
  float o0 = b2[0], o1 = b2[1], o2 = b2[2];
  const bf16_t* hp = hid + (long)node * 256;
  for (int k = 0; k < 256; ++k) {
    const float hv = (float)hp[k];
    o0 += hv * w2[k * 3 + 0];
    o1 += hv * w2[k * 3 + 1];
    o2 += hv * w2[k * 3 + 2];
  }
  out[node * 3 + 0] = o0; out[node * 3 + 1] = o1; out[node * 3 + 2] = o2;
}

static void launch_gemm(hipStream_t s, const GemmP& p) {
  dim3 g((p.M + 127) / 128, p.Ntot / 128);
  gemm_kernel<<<g, dim3(256), 0, s>>>(p);
}

extern "C" void kernel_launch(void* const* d_in, const int* in_sizes, int n_in,
                              void* d_out, int out_size, void* d_ws, size_t ws_size,
                              hipStream_t stream) {
  const float* pos      = (const float*)d_in[0];
  const float* c_in     = (const float*)d_in[1];
  const int*   at       = (const int*)d_in[2];
  const int*   ac       = (const int*)d_in[3];
  const int*   rc       = (const int*)d_in[4];
  const int*   ri       = (const int*)d_in[5];
  const int*   snd      = (const int*)d_in[6];
  const int*   rcv      = (const int*)d_in[7];
  const int*   bond     = (const int*)d_in[8];
  const float* emb_at   = (const float*)d_in[9];
  const float* emb_ac   = (const float*)d_in[10];
  const float* emb_rc   = (const float*)d_in[11];
  const float* emb_ri   = (const float*)d_in[12];
  const float* emb_bond = (const float*)d_in[13];
  const float* nenc_w1  = (const float*)d_in[14];
  const float* nenc_b1  = (const float*)d_in[15];
  const float* nenc_w2  = (const float*)d_in[16];
  const float* nenc_b2  = (const float*)d_in[17];
  const float* eenc_w1  = (const float*)d_in[18];
  const float* eenc_b1  = (const float*)d_in[19];
  const float* eenc_w2  = (const float*)d_in[20];
  const float* eenc_b2  = (const float*)d_in[21];
  const float* msg_w1   = (const float*)d_in[22];
  const float* msg_b1   = (const float*)d_in[23];
  const float* msg_w2   = (const float*)d_in[24];
  const float* msg_b2   = (const float*)d_in[25];
  const float* gate_w   = (const float*)d_in[26];
  const float* gate_b   = (const float*)d_in[27];
  const float* upd_w1   = (const float*)d_in[28];
  const float* upd_b1   = (const float*)d_in[29];
  const float* upd_w2   = (const float*)d_in[30];
  const float* upd_b2   = (const float*)d_in[31];
  const float* head_w1  = (const float*)d_in[32];
  const float* head_b1  = (const float*)d_in[33];
  const float* head_w2  = (const float*)d_in[34];
  const float* head_b2  = (const float*)d_in[35];

  const int N = in_sizes[2];
  const int E = in_sizes[6];

  char* ws = (char*)d_ws;
  size_t off = 0;
  auto alloc = [&](size_t bytes) -> char* {
    char* p = ws + off;
    off = (off + bytes + 255) & ~(size_t)255;
    return p;
  };

  bf16_t* nencW1t = (bf16_t*)alloc((size_t)256 * 128 * 2);
  bf16_t* nencW2t = (bf16_t*)alloc((size_t)128 * 256 * 2);
  bf16_t* eencW1t = (bf16_t*)alloc((size_t)256 * 96 * 2);
  bf16_t* eencW2t = (bf16_t*)alloc((size_t)128 * 256 * 2);
  bf16_t* msgW1t  = (bf16_t*)alloc((size_t)4 * 256 * 384 * 2);
  bf16_t* msgW2t  = (bf16_t*)alloc((size_t)4 * 128 * 256 * 2);
  bf16_t* updW1t  = (bf16_t*)alloc((size_t)4 * 256 * 256 * 2);
  bf16_t* updW2t  = (bf16_t*)alloc((size_t)4 * 128 * 256 * 2);
  bf16_t* headW1t = (bf16_t*)alloc((size_t)256 * 128 * 2);
  bf16_t* nemb    = (bf16_t*)alloc((size_t)N * 128 * 2);
  bf16_t* efeat   = (bf16_t*)alloc((size_t)E * 96 * 2);
  float*  cut     = (float*) alloc((size_t)E * 4);
  bf16_t* tmpE    = (bf16_t*)alloc((size_t)E * 256 * 2);
  bf16_t* tmpN    = (bf16_t*)alloc((size_t)N * 256 * 2);
  bf16_t* hbuf    = (bf16_t*)alloc((size_t)N * 128 * 2);
  bf16_t* ebuf    = (bf16_t*)alloc((size_t)E * 128 * 2);
  bf16_t* mbuf    = (bf16_t*)alloc((size_t)E * 128 * 2);
  float*  agg     = (float*) alloc((size_t)N * 128 * 4);
  bf16_t* aggbf   = (bf16_t*)alloc((size_t)N * 128 * 2);

  auto conv = [&](const float* w, bf16_t* wt, int K, int Nc, int Kpad) {
    const long tot = (long)Nc * Kpad;
    wt_conv_kernel<<<(unsigned)((tot + 255) / 256), 256, 0, stream>>>(w, wt, K, Nc, Kpad);
  };
  conv(nenc_w1, nencW1t, 128, 256, 128);
  conv(nenc_w2, nencW2t, 256, 128, 256);
  conv(eenc_w1, eencW1t, 88, 256, 96);
  conv(eenc_w2, eencW2t, 256, 128, 256);
  for (int s = 0; s < 4; ++s) {
    conv(msg_w1 + (long)s * 384 * 256, msgW1t + (long)s * 256 * 384, 384, 256, 384);
    conv(msg_w2 + (long)s * 256 * 128, msgW2t + (long)s * 128 * 256, 256, 128, 256);
    conv(upd_w1 + (long)s * 256 * 256, updW1t + (long)s * 256 * 256, 256, 256, 256);
    conv(upd_w2 + (long)s * 256 * 128, updW2t + (long)s * 128 * 256, 256, 128, 256);
  }
  conv(head_w1, headW1t, 128, 256, 128);

  node_embed_kernel<<<(unsigned)(((long)N * 128 + 255) / 256), 256, 0, stream>>>(
      at, ac, rc, ri, emb_at, emb_ac, emb_rc, emb_ri, nemb, N);
  edge_geom_kernel<<<(E + 255) / 256, 256, 0, stream>>>(
      pos, c_in, snd, rcv, bond, emb_bond, efeat, cut, E);

  // node encoder: 128 -> 256 (silu) -> 128 (rms) => h
  {
    GemmP p{}; p.a0 = nemb; p.ld0 = 128; p.k0 = 128;
    p.wt = nencW1t; p.bias = nenc_b1; p.outA = tmpN; p.ldoA = 256;
    p.M = N; p.Ntot = 256; p.flags = 1; launch_gemm(stream, p);
    GemmP q{}; q.a0 = tmpN; q.ld0 = 256; q.k0 = 256;
    q.wt = nencW2t; q.bias = nenc_b2; q.outA = hbuf; q.ldoA = 128;
    q.M = N; q.Ntot = 128; q.flags = 2; launch_gemm(stream, q);
  }
  // edge encoder: 96(pad) -> 256 (silu) -> 128 (rms) => e
  {
    GemmP p{}; p.a0 = efeat; p.ld0 = 96; p.k0 = 96;
    p.wt = eencW1t; p.bias = eenc_b1; p.outA = tmpE; p.ldoA = 256;
    p.M = E; p.Ntot = 256; p.flags = 1; launch_gemm(stream, p);
    GemmP q{}; q.a0 = tmpE; q.ld0 = 256; q.k0 = 256;
    q.wt = eencW2t; q.bias = eenc_b2; q.outA = ebuf; q.ldoA = 128;
    q.M = E; q.Ntot = 128; q.flags = 2; launch_gemm(stream, q);
  }

  for (int s = 0; s < 4; ++s) {
    // msg layer1: concat(h[snd], h[rcv], e) 384 -> 256 (silu), gather fused
    GemmP p{};
    p.a0 = hbuf; p.i0 = snd; p.ld0 = 128; p.k0 = 128;
    p.a1 = hbuf; p.i1 = rcv; p.ld1 = 128; p.k1 = 128;
    p.a2 = ebuf;             p.ld2 = 128; p.k2 = 128;
    p.wt = msgW1t + (long)s * 256 * 384; p.bias = msg_b1 + s * 256;
    p.outA = tmpE; p.ldoA = 256; p.M = E; p.Ntot = 256; p.flags = 1;
    launch_gemm(stream, p);
    // msg layer2: 256 -> 128 (rms) => m ; fused e += m
    GemmP q{}; q.a0 = tmpE; q.ld0 = 256; q.k0 = 256;
    q.wt = msgW2t + (long)s * 128 * 256; q.bias = msg_b2 + s * 128;
    q.outA = mbuf; q.ldoA = 128;
    q.outB = ebuf; q.resB = ebuf; q.ldoB = 128;
    q.M = E; q.Ntot = 128; q.flags = 2;
    launch_gemm(stream, q);

    (void)hipMemsetAsync(agg, 0, (size_t)N * 128 * 4, stream);
    gate_scatter_kernel<<<(E + 7) / 8, 256, 0, stream>>>(
        mbuf, rcv, cut, gate_w + s * 128, gate_b + s, agg, E);
    f2bf_kernel<<<(unsigned)(((long)N * 128 + 255) / 256), 256, 0, stream>>>(
        agg, aggbf, (long)N * 128);

    // update: concat(h, agg) 256 -> 256 (silu) -> 128 (rms), fused h += out
    GemmP u{};
    u.a0 = hbuf;  u.ld0 = 128; u.k0 = 128;
    u.a1 = aggbf; u.ld1 = 128; u.k1 = 128;
    u.wt = updW1t + (long)s * 256 * 256; u.bias = upd_b1 + s * 256;
    u.outA = tmpN; u.ldoA = 256; u.M = N; u.Ntot = 256; u.flags = 1;
    launch_gemm(stream, u);
    GemmP v{}; v.a0 = tmpN; v.ld0 = 256; v.k0 = 256;
    v.wt = updW2t + (long)s * 128 * 256; v.bias = upd_b2 + s * 128;
    v.outB = hbuf; v.resB = hbuf; v.ldoB = 128;
    v.M = N; v.Ntot = 128; v.flags = 2;
    launch_gemm(stream, v);
  }

  // head: 128 -> 256 (silu) -> 3
  {
    GemmP p{}; p.a0 = hbuf; p.ld0 = 128; p.k0 = 128;
    p.wt = headW1t; p.bias = head_b1; p.outA = tmpN; p.ldoA = 256;
    p.M = N; p.Ntot = 256; p.flags = 1; launch_gemm(stream, p);
    head2_kernel<<<(N + 255) / 256, 256, 0, stream>>>(
        tmpN, head_w2, head_b2, (float*)d_out, N);
  }
}
